// ForgettingAttention_13632226197676
// MI455X (gfx1250) — matrix operand order repaired and checked by reading, not compiled, not run on hardware
//
#include <hip/hip_runtime.h>

// ---------------------------------------------------------------------------
// Forgetting-gate attention (FoX) for MI455X / gfx1250, wave32, WMMA bf16.
// Whole working set (~80MB) is L2-resident (192MB), so all matmul FLOPs run on
// v_wmma_f32_16x16x32_bf16 fed by direct global b128 loads (no LDS staging
// needed for GEMM). Round-1 disasm confirmed load/WMMA software pipelining
// with partial s_wait_loadcnt; this round raises arithmetic intensity:
//   - GEMM wave tile 32x64 -> 64x64 (1.5 -> 1.0 b128 loads per WMMA)
//   - attention 16 -> 32 query rows/wave (K/V fragments amortized 2x)
// ---------------------------------------------------------------------------

#define N_EMBD   1024
#define N_HEADS  16
#define HEAD_D   64
#define TLEN     2048
#define BATCH    2
#define QDIM     1024
#define EPSV     1e-6f

typedef __attribute__((ext_vector_type(16))) __bf16        v16bf;
typedef __attribute__((ext_vector_type(8)))  float         v8f;
typedef __attribute__((ext_vector_type(4)))  unsigned int  v4u;
typedef __attribute__((ext_vector_type(4)))  float         v4f;

union FragU { v16bf bf; v4u q[2]; };

// fp32 -> bf16 round-to-nearest-even (bit ops only)
__device__ __forceinline__ unsigned short f2bf(float x) {
  unsigned int u = __builtin_bit_cast(unsigned int, x);
  u += 0x7FFFu + ((u >> 16) & 1u);
  return (unsigned short)(u >> 16);
}

__device__ __forceinline__ v8f wmma_bf16(v16bf a, v16bf b, v8f c) {
  return __builtin_amdgcn_wmma_f32_16x16x32_bf16(false, a, false, b, (short)0, c,
                                                 false, false);
}

// A-matrix fragment (16x32, 16-bit): lane m=L%16, half=L/16.
// ISA: VGPR0-3 hold K = half*8 + 0..7 ; VGPR4-7 hold K = 16 + half*8 + 0..7.
__device__ __forceinline__ v16bf load_fragA(const unsigned short* base, int ld,
                                            int m, int half) {
  FragU f;
  const unsigned short* p = base + (size_t)m * ld + half * 8;
  f.q[0] = *reinterpret_cast<const v4u*>(p);
  f.q[1] = *reinterpret_cast<const v4u*>(p + 16);
  return f.bf;
}

// B-matrix fragment (32x16, 16-bit) from B^T rows: K = half*8 + 0..7 then
// K = 16 + half*8 + 0..7 (same K order as A fragment).
__device__ __forceinline__ v16bf load_fragB(const unsigned short* base, int ld,
                                            int n, int half) {
  FragU f;
  const unsigned short* p = base + (size_t)n * ld + half * 8;
  f.q[0] = *reinterpret_cast<const v4u*>(p);
  f.q[1] = *reinterpret_cast<const v4u*>(p + 16);
  return f.bf;
}

// Reductions across a 16-lane group (xor masks <= 8 stay inside the group).
__device__ __forceinline__ float grp16_sum(float v) {
  v += __shfl_xor(v, 1); v += __shfl_xor(v, 2);
  v += __shfl_xor(v, 4); v += __shfl_xor(v, 8);
  return v;
}
__device__ __forceinline__ float grp16_max(float v) {
  v = fmaxf(v, __shfl_xor(v, 1)); v = fmaxf(v, __shfl_xor(v, 2));
  v = fmaxf(v, __shfl_xor(v, 4)); v = fmaxf(v, __shfl_xor(v, 8));
  return v;
}

// ---------------------------------------------------------------------------
// GEMM: C[M,N] (fp32) = A[M,K] (bf16 row-major) @ Bt[N,K]^T (bf16).
// 8 waves/block; each wave owns a 64x64 tile (4x4 WMMA tiles), K-step 32.
// 16 WMMAs per 16 b128 loads per K-step. Block tile 256x128.
// ---------------------------------------------------------------------------
__global__ __launch_bounds__(256) void gemm_bf16_nt(
    const unsigned short* __restrict__ A, const unsigned short* __restrict__ Bt,
    float* __restrict__ C, int M, int N, int K) {
  const int lane = threadIdx.x & 31;
  const int wave = threadIdx.x >> 5;
  const int m = lane & 15, half = lane >> 4;
  const int nBlk = N >> 7;                 // block tile: 256 (M) x 128 (N)
  const int bm = blockIdx.x / nBlk;
  const int bn = blockIdx.x % nBlk;
  const int row0 = bm * 256 + (wave >> 1) * 64;
  const int col0 = bn * 128 + (wave & 1) * 64;
  (void)M;

  v8f acc[4][4] = {};
  for (int k0 = 0; k0 < K; k0 += 32) {
    v16bf a[4], b[4];
    #pragma unroll
    for (int i = 0; i < 4; ++i)
      a[i] = load_fragA(A + (size_t)(row0 + 16 * i) * K + k0, K, m, half);
    #pragma unroll
    for (int j = 0; j < 4; ++j)
      b[j] = load_fragB(Bt + (size_t)(col0 + 16 * j) * K + k0, K, m, half);
    #pragma unroll
    for (int i = 0; i < 4; ++i)
      #pragma unroll
      for (int j = 0; j < 4; ++j)
        acc[i][j] = wmma_bf16(a[i], b[j], acc[i][j]);
  }
  // C/D layout: VGPR r holds row (r + 8*half), column = lane%16.
  #pragma unroll
  for (int i = 0; i < 4; ++i)
    #pragma unroll
    for (int j = 0; j < 4; ++j)
      #pragma unroll
      for (int r = 0; r < 8; ++r) {
        int row = row0 + i * 16 + r + 8 * half;
        int col = col0 + j * 16 + m;
        C[(size_t)row * N + col] = acc[i][j][r];
      }
}

// ---------------------------------------------------------------------------
// Flash attention with forgetting-gate decay. One wave = 32 query rows
// (2 row-tiles), s-step 32 keys. Per-row constant cum[t] cancels in softmax,
// so decay is only a per-COLUMN bias (-cum[s]); 1/sqrt(64) pre-folded into q.
// Q,K: [B*H, T, 64] bf16; V^T: [B*H, 64, T] bf16.
// ---------------------------------------------------------------------------
__global__ __launch_bounds__(256) void attn_fox_kernel(
    const unsigned short* __restrict__ Q, const unsigned short* __restrict__ Kh,
    const unsigned short* __restrict__ Vt, const float* __restrict__ cum,
    float* __restrict__ Oa) {
  __shared__ __align__(16) unsigned short ptile[8][32 * 32];  // per-wave P tiles
  const int lane = threadIdx.x & 31;
  const int wave = threadIdx.x >> 5;
  const int m = lane & 15, half = lane >> 4;
  const int w = blockIdx.x * 8 + wave;
  const int qt = w & ((TLEN / 32) - 1);     // 64 q-tiles of 32 rows
  const int bh = w >> 6;
  const int qbase = qt * 32;
  const unsigned short* qb = Q  + (size_t)bh * TLEN * HEAD_D;
  const unsigned short* kb = Kh + (size_t)bh * TLEN * HEAD_D;
  const unsigned short* vb = Vt + (size_t)bh * HEAD_D * TLEN;
  const float* cb = cum + (size_t)bh * TLEN;

  v16bf qf[2][2];
  #pragma unroll
  for (int i = 0; i < 2; ++i) {
    qf[i][0] = load_fragA(qb + (size_t)(qbase + 16 * i) * HEAD_D +  0, HEAD_D, m, half);
    qf[i][1] = load_fragA(qb + (size_t)(qbase + 16 * i) * HEAD_D + 32, HEAD_D, m, half);
  }

  v8f o[2][4] = {};
  float Mr[2][8], Lr[2][8];
  #pragma unroll
  for (int i = 0; i < 2; ++i)
    #pragma unroll
    for (int r = 0; r < 8; ++r) { Mr[i][r] = -__builtin_inff(); Lr[i][r] = 0.0f; }

  const int tlast = qbase + 31;
  for (int s0 = 0; s0 <= tlast; s0 += 32) {
    v16bf kB[2][2];
    #pragma unroll
    for (int n = 0; n < 2; ++n) {
      kB[n][0] = load_fragB(kb + (size_t)(s0 + 16 * n) * HEAD_D +  0, HEAD_D, m, half);
      kB[n][1] = load_fragB(kb + (size_t)(s0 + 16 * n) * HEAD_D + 32, HEAD_D, m, half);
    }
    v8f S[2][2] = {};
    #pragma unroll
    for (int i = 0; i < 2; ++i)
      #pragma unroll
      for (int n = 0; n < 2; ++n) {
        S[i][n] = wmma_bf16(qf[i][0], kB[n][0], S[i][n]);
        S[i][n] = wmma_bf16(qf[i][1], kB[n][1], S[i][n]);
      }

    const int c0 = s0 + m, c1 = s0 + 16 + m;  // key index per lane
    const float nc0 = -cb[c0];
    const float nc1 = -cb[c1];
    #pragma unroll
    for (int i = 0; i < 2; ++i) {
      #pragma unroll
      for (int r = 0; r < 8; ++r) {
        const int t = qbase + 16 * i + r + 8 * half;   // query row of VGPR r
        float v0 = (c0 <= t) ? (S[i][0][r] + nc0) : -__builtin_inff();
        float v1 = (c1 <= t) ? (S[i][1][r] + nc1) : -__builtin_inff();
        float mx = grp16_max(fmaxf(v0, v1));
        float newM = fmaxf(Mr[i][r], mx);
        float alpha = (newM > -__builtin_inff()) ? expf(Mr[i][r] - newM) : 1.0f;
        float p0 = (v0 > -__builtin_inff()) ? expf(v0 - newM) : 0.0f;
        float p1 = (v1 > -__builtin_inff()) ? expf(v1 - newM) : 0.0f;
        float ps = grp16_sum(p0 + p1);
        Lr[i][r] = Lr[i][r] * alpha + ps;
        Mr[i][r] = newM;
        #pragma unroll
        for (int j = 0; j < 4; ++j) o[i][j][r] *= alpha;
        const int prow = 16 * i + r + 8 * half;
        ptile[wave][prow * 32 + m]      = f2bf(p0);
        ptile[wave][prow * 32 + 16 + m] = f2bf(p1);
      }
    }
    // Wave-private LDS RAW: DS ops are in-order per wave; wait then reload as
    // A-fragments (transposes C/D layout -> A layout).
    asm volatile("s_wait_dscnt 0" ::: "memory");
    const v16bf pf0 = load_fragA(&ptile[wave][0],       32, m, half);
    const v16bf pf1 = load_fragA(&ptile[wave][16 * 32], 32, m, half);
    #pragma unroll
    for (int j = 0; j < 4; ++j) {
      v16bf vf = load_fragB(vb + (size_t)(j * 16) * TLEN + s0, TLEN, m, half);
      o[0][j] = wmma_bf16(pf0, vf, o[0][j]);
      o[1][j] = wmma_bf16(pf1, vf, o[1][j]);
    }
  }

  const int b = bh >> 4, h = bh & (N_HEADS - 1);
  #pragma unroll
  for (int i = 0; i < 2; ++i)
    #pragma unroll
    for (int r = 0; r < 8; ++r) {
      const float inv = 1.0f / Lr[i][r];
      const int t = qbase + 16 * i + r + 8 * half;
      const size_t rowoff = (((size_t)b * TLEN + t) * N_HEADS + h) * HEAD_D;
      #pragma unroll
      for (int j = 0; j < 4; ++j) Oa[rowoff + j * 16 + m] = o[i][j][r] * inv;
    }
}

// ---------------------------------------------------------------------------
// Elementwise / prep kernels
// ---------------------------------------------------------------------------
__global__ __launch_bounds__(256) void pack_bf16_kernel(
    const float* __restrict__ in, unsigned short* __restrict__ out, int n) {
  int i = blockIdx.x * 256 + threadIdx.x;
  if (i < n) out[i] = f2bf(in[i]);
}

__global__ __launch_bounds__(256) void transpose_bf16_kernel(
    const float* __restrict__ in, unsigned short* __restrict__ out, int R, int Cc) {
  int i = blockIdx.x * 256 + threadIdx.x;
  if (i < R * Cc) {
    int r = i / Cc, c = i - r * Cc;
    out[(size_t)c * R + r] = f2bf(in[i]);
  }
}

// fgate logits + log-sigmoid; thread = (bt, h). Wf column reads coalesce.
__global__ __launch_bounds__(256) void fgate_kernel(
    const float* __restrict__ x, const float* __restrict__ Wf,
    const float* __restrict__ bfv, float* __restrict__ lf) {
  int idx = blockIdx.x * 256 + threadIdx.x;
  int h = idx & (N_HEADS - 1);
  int bt = idx >> 4;
  const float* xr = x + (size_t)bt * N_EMBD;
  float acc = 0.0f;
  for (int i = 0; i < N_EMBD; ++i) acc += xr[i] * Wf[i * N_HEADS + h];
  float z = acc + bfv[h];
  float ls = fminf(z, 0.0f) - log1pf(expf(-fabsf(z)));
  int b = bt / TLEN, t = bt - b * TLEN;
  lf[((size_t)(b * N_HEADS + h)) * TLEN + t] = ls;
}

__global__ void cumsum_kernel(const float* __restrict__ lf, float* __restrict__ cum) {
  int bh = threadIdx.x;  // 32 sequences, serial scan (tiny)
  const float* p = lf + (size_t)bh * TLEN;
  float* q = cum + (size_t)bh * TLEN;
  float run = 0.0f;
  for (int t = 0; t < TLEN; ++t) { run += p[t]; q[t] = run; }
}

// per-head RMSNorm on q (softmax scale folded in) -> [B*H, T, 64] bf16
__global__ __launch_bounds__(256) void qnorm_kernel(
    const float* __restrict__ raw, const float* __restrict__ w,
    unsigned short* __restrict__ outbh, float scale) {
  int bt = blockIdx.x, tid = threadIdx.x, c = tid * 4;
  v4f xv = *reinterpret_cast<const v4f*>(raw + (size_t)bt * QDIM + c);
  float ss = grp16_sum(xv.x * xv.x + xv.y * xv.y + xv.z * xv.z + xv.w * xv.w);
  float rinv = rsqrtf(ss * (1.0f / HEAD_D) + EPSV);
  int h = tid >> 4;
  int b = bt / TLEN, t = bt - b * TLEN;
  size_t o = (((size_t)(b * N_HEADS + h)) * TLEN + t) * HEAD_D + (c & 63);
  outbh[o + 0] = f2bf(xv.x * rinv * w[c + 0] * scale);
  outbh[o + 1] = f2bf(xv.y * rinv * w[c + 1] * scale);
  outbh[o + 2] = f2bf(xv.z * rinv * w[c + 2] * scale);
  outbh[o + 3] = f2bf(xv.w * rinv * w[c + 3] * scale);
}

// causal depthwise conv(K=4) + SiLU + per-head RMSNorm -> k [B*H, T, 64] bf16
__global__ __launch_bounds__(256) void kconv_norm_kernel(
    const float* __restrict__ raw, const float* __restrict__ cw,
    const float* __restrict__ w, unsigned short* __restrict__ outbh) {
  int bt = blockIdx.x, tid = threadIdx.x, c = tid * 4;
  int b = bt / TLEN, t = bt - b * TLEN;
  float a0 = 0, a1 = 0, a2 = 0, a3 = 0;
  #pragma unroll
  for (int j = 0; j < 4; ++j) {
    int tt = t - 3 + j;
    if (tt >= 0) {
      v4f xv = *reinterpret_cast<const v4f*>(raw + ((size_t)b * TLEN + tt) * QDIM + c);
      a0 += xv.x * cw[(c + 0) * 4 + j];
      a1 += xv.y * cw[(c + 1) * 4 + j];
      a2 += xv.z * cw[(c + 2) * 4 + j];
      a3 += xv.w * cw[(c + 3) * 4 + j];
    }
  }
  float y0 = a0 / (1.0f + expf(-a0));
  float y1 = a1 / (1.0f + expf(-a1));
  float y2 = a2 / (1.0f + expf(-a2));
  float y3 = a3 / (1.0f + expf(-a3));
  float ss = grp16_sum(y0 * y0 + y1 * y1 + y2 * y2 + y3 * y3);
  float rinv = rsqrtf(ss * (1.0f / HEAD_D) + EPSV);
  int h = tid >> 4;
  size_t o = (((size_t)(b * N_HEADS + h)) * TLEN + t) * HEAD_D + (c & 63);
  outbh[o + 0] = f2bf(y0 * rinv * w[c + 0]);
  outbh[o + 1] = f2bf(y1 * rinv * w[c + 1]);
  outbh[o + 2] = f2bf(y2 * rinv * w[c + 2]);
  outbh[o + 3] = f2bf(y3 * rinv * w[c + 3]);
}

// causal depthwise conv(K=4) + SiLU on v -> transposed V^T [B*H, 64, T] bf16
__global__ __launch_bounds__(256) void vconv_kernel(
    const float* __restrict__ raw, const float* __restrict__ cw,
    unsigned short* __restrict__ vt) {
  int bt = blockIdx.x, tid = threadIdx.x, c = tid * 4;
  int b = bt / TLEN, t = bt - b * TLEN;
  float a0 = 0, a1 = 0, a2 = 0, a3 = 0;
  #pragma unroll
  for (int j = 0; j < 4; ++j) {
    int tt = t - 3 + j;
    if (tt >= 0) {
      v4f xv = *reinterpret_cast<const v4f*>(raw + ((size_t)b * TLEN + tt) * QDIM + c);
      a0 += xv.x * cw[(c + 0) * 4 + j];
      a1 += xv.y * cw[(c + 1) * 4 + j];
      a2 += xv.z * cw[(c + 2) * 4 + j];
      a3 += xv.w * cw[(c + 3) * 4 + j];
    }
  }
  int h = tid >> 4;
  int d = c & 63;
  size_t o = ((size_t)(b * N_HEADS + h) * HEAD_D + d) * TLEN + t;
  vt[o + 0 * TLEN] = f2bf(a0 / (1.0f + expf(-a0)));
  vt[o + 1 * TLEN] = f2bf(a1 / (1.0f + expf(-a1)));
  vt[o + 2 * TLEN] = f2bf(a2 / (1.0f + expf(-a2)));
  vt[o + 3 * TLEN] = f2bf(a3 / (1.0f + expf(-a3)));
}

// per-head RMSNorm(attn out) * onw * swish(g) -> bf16 rows for final GEMM
__global__ __launch_bounds__(256) void gate_kernel(
    const float* __restrict__ oatt, const float* __restrict__ graw,
    const float* __restrict__ onw, unsigned short* __restrict__ attin) {
  int bt = blockIdx.x, tid = threadIdx.x, c = tid * 4;
  v4f ov = *reinterpret_cast<const v4f*>(oatt + (size_t)bt * QDIM + c);
  v4f gv = *reinterpret_cast<const v4f*>(graw + (size_t)bt * QDIM + c);
  float ss = grp16_sum(ov.x * ov.x + ov.y * ov.y + ov.z * ov.z + ov.w * ov.w);
  float rinv = rsqrtf(ss * (1.0f / HEAD_D) + EPSV);
  int d = c & 63;
  float s0 = gv.x / (1.0f + expf(-gv.x));
  float s1 = gv.y / (1.0f + expf(-gv.y));
  float s2 = gv.z / (1.0f + expf(-gv.z));
  float s3 = gv.w / (1.0f + expf(-gv.w));
  size_t o = (size_t)bt * QDIM + c;
  attin[o + 0] = f2bf(ov.x * rinv * onw[d + 0] * s0);
  attin[o + 1] = f2bf(ov.y * rinv * onw[d + 1] * s1);
  attin[o + 2] = f2bf(ov.z * rinv * onw[d + 2] * s2);
  attin[o + 3] = f2bf(ov.w * rinv * onw[d + 3] * s3);
}

// ---------------------------------------------------------------------------
extern "C" void kernel_launch(void* const* d_in, const int* in_sizes, int n_in,
                              void* d_out, int out_size, void* d_ws, size_t ws_size,
                              hipStream_t stream) {
  (void)in_sizes; (void)n_in; (void)out_size; (void)ws_size;
  const float* x   = (const float*)d_in[0];
  const float* Wq  = (const float*)d_in[1];
  const float* Wk  = (const float*)d_in[2];
  const float* Wv  = (const float*)d_in[3];
  const float* Wo  = (const float*)d_in[4];
  const float* Wf  = (const float*)d_in[5];
  const float* bfv = (const float*)d_in[6];
  const float* Wg  = (const float*)d_in[7];
  const float* qnw = (const float*)d_in[8];
  const float* knw = (const float*)d_in[9];
  const float* onw = (const float*)d_in[10];
  const float* kcw = (const float*)d_in[11];
  const float* vcw = (const float*)d_in[12];

  char* ws = (char*)d_ws;
  size_t off = 0;
  auto alloc = [&](size_t bytes) -> char* {
    char* p = ws + off;
    off += (bytes + 255) & ~(size_t)255;
    return p;
  };
  const int BT = BATCH * TLEN;  // 4096
  unsigned short* xb  = (unsigned short*)alloc((size_t)BT * QDIM * 2);
  unsigned short* WqT = (unsigned short*)alloc((size_t)N_EMBD * QDIM * 2);
  unsigned short* WkT = (unsigned short*)alloc((size_t)N_EMBD * QDIM * 2);
  unsigned short* WvT = (unsigned short*)alloc((size_t)N_EMBD * QDIM * 2);
  unsigned short* WgT = (unsigned short*)alloc((size_t)N_EMBD * QDIM * 2);
  unsigned short* WoT = (unsigned short*)alloc((size_t)QDIM * N_EMBD * 2);
  float* raw = (float*)alloc((size_t)BT * QDIM * 4);          // q/k/v/g raw (reused)
  unsigned short* qbh = (unsigned short*)alloc((size_t)BATCH * N_HEADS * TLEN * HEAD_D * 2);
  unsigned short* kbh = (unsigned short*)alloc((size_t)BATCH * N_HEADS * TLEN * HEAD_D * 2);
  unsigned short* vtb = (unsigned short*)alloc((size_t)BATCH * N_HEADS * HEAD_D * TLEN * 2);
  float* lf   = (float*)alloc((size_t)BATCH * N_HEADS * TLEN * 4);
  float* cum  = (float*)alloc((size_t)BATCH * N_HEADS * TLEN * 4);
  float* oatt = (float*)alloc((size_t)BT * QDIM * 4);
  unsigned short* attin = xb;   // reuse: x_bf16 dead after the g projection

  // prep: bf16 x, bf16-transposed weights, forget-gate cumsum
  pack_bf16_kernel<<<(BT * QDIM) / 256, 256, 0, stream>>>(x, xb, BT * QDIM);
  transpose_bf16_kernel<<<(N_EMBD * QDIM) / 256, 256, 0, stream>>>(Wq, WqT, N_EMBD, QDIM);
  transpose_bf16_kernel<<<(N_EMBD * QDIM) / 256, 256, 0, stream>>>(Wk, WkT, N_EMBD, QDIM);
  transpose_bf16_kernel<<<(N_EMBD * QDIM) / 256, 256, 0, stream>>>(Wv, WvT, N_EMBD, QDIM);
  transpose_bf16_kernel<<<(N_EMBD * QDIM) / 256, 256, 0, stream>>>(Wg, WgT, N_EMBD, QDIM);
  transpose_bf16_kernel<<<(QDIM * N_EMBD) / 256, 256, 0, stream>>>(Wo, WoT, QDIM, N_EMBD);
  fgate_kernel<<<(BT * N_HEADS) / 256, 256, 0, stream>>>(x, Wf, bfv, lf);
  cumsum_kernel<<<1, BATCH * N_HEADS, 0, stream>>>(lf, cum);

  const int gemmGrid = (BT / 256) * (QDIM / 128);  // 128 blocks
  // q
  gemm_bf16_nt<<<gemmGrid, 256, 0, stream>>>(xb, WqT, raw, BT, QDIM, N_EMBD);
  qnorm_kernel<<<BT, 256, 0, stream>>>(raw, qnw, qbh, 0.125f);  // fold 1/sqrt(64)
  // k
  gemm_bf16_nt<<<gemmGrid, 256, 0, stream>>>(xb, WkT, raw, BT, QDIM, N_EMBD);
  kconv_norm_kernel<<<BT, 256, 0, stream>>>(raw, kcw, knw, kbh);
  // v
  gemm_bf16_nt<<<gemmGrid, 256, 0, stream>>>(xb, WvT, raw, BT, QDIM, N_EMBD);
  vconv_kernel<<<BT, 256, 0, stream>>>(raw, vcw, vtb);
  // g (stays in raw until the gate kernel)
  gemm_bf16_nt<<<gemmGrid, 256, 0, stream>>>(xb, WgT, raw, BT, QDIM, N_EMBD);
  // attention: B*H*(T/32) waves / 8 per block = 256 blocks
  attn_fox_kernel<<<(BATCH * N_HEADS * (TLEN / 32)) / 8, 256, 0, stream>>>(
      qbh, kbh, vtb, cum, oatt);
  // gated norm + output projection
  gate_kernel<<<BT, 256, 0, stream>>>(oatt, raw, onw, attin);
  gemm_bf16_nt<<<gemmGrid, 256, 0, stream>>>(attin, WoT, (float*)d_out, BT, N_EMBD, QDIM);
}
